// DemonstrationAttentionQModel_64175401337268
// MI455X (gfx1250) — matrix operand order repaired and checked by reading, not compiled, not run on hardware
//
#include <hip/hip_runtime.h>
#include <hip/hip_bf16.h>
#include <cstdint>
#include <cstddef>

// ---------------------------------------------------------------------------
// Model dims (from reference): T=40, B=64, L=80, D_IN=256, H=1024, A=18
// ---------------------------------------------------------------------------
#define TT   40
#define BB   64
#define LL   80
#define DIN  256
#define HH   1024
#define AA   18

// GEMM workgroup shape: 4 waves x 16 cols = 64 output columns per WG.
#define GWAVES   4
#define GTHREADS 128
#define GCOLS    64

typedef __bf16 v16bf __attribute__((ext_vector_type(16)));
typedef float  v8f   __attribute__((ext_vector_type(8)));

__device__ __forceinline__ float sigmoidf_(float x) { return 1.0f / (1.0f + __expf(-x)); }

// Branch-free tanh: one v_exp_f32 + a few VALU (avoids libm slow path).
__device__ __forceinline__ float tanhf_(float x)
{
    float ax = fabsf(x);
    float t  = __expf(-2.0f * ax);
    float r  = (1.0f - t) / (1.0f + t);
    return copysignf(r, x);
}

// ---------------------------------------------------------------------------
// Pack a row-major fp32 weight W[N][K] (used as x @ W^T) into bf16 WMMA
// B-fragment order: tiles of 32(K) x 16(N); within a tile, 32 lanes x 16
// contiguous halfs. lane<16: N=lane, K=e ; lane>=16: N=lane-16, K=e+16.
// Tile order: nt-major, kt-minor  =>  tile = nt*KT + kt.
// ---------------------------------------------------------------------------
__global__ void pack_b_kernel(const float* __restrict__ W, __hip_bfloat16* __restrict__ out,
                              int K, int N)
{
    const int KT = K >> 5;
    const size_t total = (size_t)K * (size_t)N;
    for (size_t idx = (size_t)blockIdx.x * blockDim.x + threadIdx.x; idx < total;
         idx += (size_t)gridDim.x * blockDim.x) {
        int e    = (int)(idx & 15);
        int lane = (int)((idx >> 4) & 31);
        size_t tile = idx >> 9;
        int kt = (int)(tile % KT);
        int nt = (int)(tile / KT);
        int n = nt * 16 + (lane & 15);
        int k = kt * 32 + ((lane < 16) ? e : e + 16);
        out[idx] = __float2bfloat16(W[(size_t)n * K + k]);
    }
}

// ---------------------------------------------------------------------------
// Stage one 64x32 fp32 activation chunk into LDS as bf16 A fragments.
// A-fragment layout used here (contiguous k, same as pack_b): lane = (m%16) + 16*(k>=16),
// element e = k%16.
// ---------------------------------------------------------------------------
__device__ __forceinline__ void stage_A_chunk(__hip_bfloat16* ldsA,
                                              const float* __restrict__ src,
                                              int sld, int kb, int tid)
{
    for (int i = tid; i < 2048; i += GTHREADS) {
        int m = i >> 5, k = i & 31;
        float v = src[(size_t)m * sld + kb + k];
        int hi = (k >= 16) ? 16 : 0;
        int e  = k & 15;
        ldsA[((((m >> 4) << 5) + (m & 15) + hi) << 4) + e] = __float2bfloat16(v);
    }
}

// ---------------------------------------------------------------------------
// Fused LSTM step: G = X @ Wih^T + Hin @ Whh^T + bih + bhh ; cell update.
// Grid: 16 WGs (each owns 64 hidden columns). 128 threads = 4 waves; each
// wave owns a 16-wide column strip and computes all 4 gate strips for it, so
// the i/f/g/o combine is lane-local. Hout is double-buffered by the host.
// encOut (optional) receives h for the encoder output sequence.
// ---------------------------------------------------------------------------
__global__ void __launch_bounds__(GTHREADS)
lstm_step_kernel(const float* __restrict__ X, int ldx, int Kx,
                 const float* __restrict__ Hin,
                 const __hip_bfloat16* __restrict__ WihP,
                 const __hip_bfloat16* __restrict__ WhhP,
                 const float* __restrict__ bih, const float* __restrict__ bhh,
                 float* __restrict__ Hout, float* __restrict__ Cst,
                 float* __restrict__ encOut, int ldencB)
{
    __shared__ __align__(32) __hip_bfloat16 ldsA[2048];
    const int tid  = threadIdx.x;
    const int lane = tid & 31;
    const int wave = tid >> 5;
    const int ncol = blockIdx.x * GCOLS + wave * 16;
    const int KTx  = Kx >> 5;
    const int KCT  = KTx + (HH >> 5);

    v8f acc[4][4];
    #pragma unroll
    for (int g = 0; g < 4; ++g)
        #pragma unroll
        for (int mt = 0; mt < 4; ++mt)
            #pragma unroll
            for (int r = 0; r < 8; ++r) acc[g][mt][r] = 0.f;

    for (int kc = 0; kc < KCT; ++kc) {
        const float* src; int sld, kb; const __hip_bfloat16* Bp; int kt, KT;
        if (kc < KTx) { src = X;   sld = ldx; kb = kc << 5;          Bp = WihP; kt = kc;        KT = KTx; }
        else          { src = Hin; sld = HH;  kb = (kc - KTx) << 5;  Bp = WhhP; kt = kc - KTx;  KT = HH >> 5; }
        __syncthreads();
        stage_A_chunk(ldsA, src, sld, kb, tid);
        __syncthreads();

        v16bf bf[4];
        #pragma unroll
        for (int g = 0; g < 4; ++g) {
            int nt = (g << 6) + (ncol >> 4);
            const __hip_bfloat16* bp = Bp + ((((size_t)(nt * KT + kt) << 5) + lane) << 4);
            bf[g] = *reinterpret_cast<const v16bf*>(bp);
            __builtin_prefetch(bp + 512, 0, 3);   // next K-chunk of this strip
        }
        #pragma unroll
        for (int mt = 0; mt < 4; ++mt) {
            v16bf af = *reinterpret_cast<const v16bf*>(&ldsA[((mt << 5) + lane) << 4]);
            #pragma unroll
            for (int g = 0; g < 4; ++g)
                acc[g][mt] = __builtin_amdgcn_wmma_f32_16x16x32_bf16(
                    false, af, false, bf[g], (short)0, acc[g][mt], false, false);
        }
    }

    const int n  = ncol + (lane & 15);
    const float bI = bih[n]          + bhh[n];
    const float bF = bih[HH + n]     + bhh[HH + n];
    const float bG = bih[2 * HH + n] + bhh[2 * HH + n];
    const float bO = bih[3 * HH + n] + bhh[3 * HH + n];
    const int mhi = (lane >= 16) ? 8 : 0;
    #pragma unroll
    for (int mt = 0; mt < 4; ++mt) {
        #pragma unroll
        for (int r = 0; r < 8; ++r) {
            int m = (mt << 4) + r + mhi;
            size_t off = (size_t)m * HH + n;
            float iv = sigmoidf_(acc[0][mt][r] + bI);
            float fv = sigmoidf_(acc[1][mt][r] + bF);
            float gv = tanhf_  (acc[2][mt][r] + bG);
            float ov = sigmoidf_(acc[3][mt][r] + bO);
            float cn = fv * Cst[off] + iv * gv;
            float hn = ov * tanhf_(cn);
            Cst[off]  = cn;
            Hout[off] = hn;
            if (encOut) encOut[(size_t)m * ldencB + n] = hn;
        }
    }
}

// ---------------------------------------------------------------------------
// Generic 2-segment GEMM:  C[64x1024] = [A0|A1] @ Bp^T + bias (opt ReLU).
// Bp packed as a single (K0+K1) x 1024 matrix. Grid: 16 WGs x 128 threads.
// ---------------------------------------------------------------------------
template <bool RELU>
__global__ void __launch_bounds__(GTHREADS)
gemm2_kernel(const float* __restrict__ A0, int ld0, int K0,
             const float* __restrict__ A1, int ld1, int K1,
             const __hip_bfloat16* __restrict__ Bp,
             const float* __restrict__ bias,
             float* __restrict__ C)
{
    __shared__ __align__(32) __hip_bfloat16 ldsA[2048];
    const int tid  = threadIdx.x;
    const int lane = tid & 31;
    const int wave = tid >> 5;
    const int ncol = blockIdx.x * GCOLS + wave * 16;
    const int KT0 = K0 >> 5, KCT = KT0 + (K1 >> 5);

    v8f acc[4];
    #pragma unroll
    for (int mt = 0; mt < 4; ++mt)
        #pragma unroll
        for (int r = 0; r < 8; ++r) acc[mt][r] = 0.f;

    for (int kc = 0; kc < KCT; ++kc) {
        const float* src; int sld, kb;
        if (kc < KT0) { src = A0; sld = ld0; kb = kc << 5; }
        else          { src = A1; sld = ld1; kb = (kc - KT0) << 5; }
        __syncthreads();
        stage_A_chunk(ldsA, src, sld, kb, tid);
        __syncthreads();

        int nt = ncol >> 4;
        const __hip_bfloat16* bp = Bp + ((((size_t)(nt * KCT + kc) << 5) + lane) << 4);
        v16bf bf = *reinterpret_cast<const v16bf*>(bp);
        __builtin_prefetch(bp + 512, 0, 3);
        #pragma unroll
        for (int mt = 0; mt < 4; ++mt) {
            v16bf af = *reinterpret_cast<const v16bf*>(&ldsA[((mt << 5) + lane) << 4]);
            acc[mt] = __builtin_amdgcn_wmma_f32_16x16x32_bf16(
                false, af, false, bf, (short)0, acc[mt], false, false);
        }
    }

    const int n  = ncol + (lane & 15);
    const float bv = bias[n];
    const int mhi = (lane >= 16) ? 8 : 0;
    #pragma unroll
    for (int mt = 0; mt < 4; ++mt)
        #pragma unroll
        for (int r = 0; r < 8; ++r) {
            int m = (mt << 4) + r + mhi;
            float v = acc[mt][r] + bv;
            if (RELU) v = fmaxf(v, 0.f);
            C[(size_t)m * HH + n] = v;
        }
}

// ---------------------------------------------------------------------------
// score_d[b,l] = enc[b,l,:] . Wd + attn_b     (one block per (b,l))
// ---------------------------------------------------------------------------
__global__ void score_d_kernel(const float* __restrict__ enc,
                               const float* __restrict__ attnW,
                               const float* __restrict__ attnB,
                               float* __restrict__ score_d)
{
    const int bl = blockIdx.x, tid = threadIdx.x;
    __shared__ float red[128];
    const float* e = enc + (size_t)bl * HH;
    float p = 0.f;
    for (int n = tid; n < HH; n += 128) p += e[n] * attnW[n];
    red[tid] = p; __syncthreads();
    for (int s = 64; s > 0; s >>= 1) { if (tid < s) red[tid] += red[tid + s]; __syncthreads(); }
    if (tid == 0) score_d[bl] = red[0] + attnB[0];
}

// ---------------------------------------------------------------------------
// Per-step attention: scores = score_d + h.Wh, masked softmax over L=80,
// attn[b,:] = sum_l w[l]*enc[b,l,:].   One block per batch row.
// ---------------------------------------------------------------------------
__global__ void attn_kernel(const float* __restrict__ h,
                            const float* __restrict__ enc,
                            const float* __restrict__ score_d,
                            const float* __restrict__ attnW,
                            const int*   __restrict__ lens,
                            float* __restrict__ out)
{
    const int b = blockIdx.x, tid = threadIdx.x;   // 128 threads
    __shared__ float red[128];
    __shared__ float w[LL];

    const float* hb = h + (size_t)b * HH;
    const float* Wh = attnW + HH;
    float p = 0.f;
    for (int n = tid; n < HH; n += 128) p += hb[n] * Wh[n];
    red[tid] = p; __syncthreads();
    for (int s = 64; s > 0; s >>= 1) { if (tid < s) red[tid] += red[tid + s]; __syncthreads(); }
    const float hdot = red[0];
    __syncthreads();

    const int len = lens[b];
    float sc = -3.0e38f;
    if (tid < LL) {
        sc = score_d[b * LL + tid] + hdot;
        if (tid >= len) sc = -3.0e38f;
        w[tid] = sc;
    }
    red[tid] = (tid < LL) ? sc : -3.0e38f; __syncthreads();
    for (int s = 64; s > 0; s >>= 1) { if (tid < s) red[tid] = fmaxf(red[tid], red[tid + s]); __syncthreads(); }
    const float mx = red[0];
    __syncthreads();

    float ev = 0.f;
    if (tid < LL) { ev = __expf(w[tid] - mx); w[tid] = ev; }
    red[tid] = (tid < LL) ? ev : 0.f; __syncthreads();
    for (int s = 64; s > 0; s >>= 1) { if (tid < s) red[tid] += red[tid + s]; __syncthreads(); }
    const float inv = 1.0f / red[0];
    __syncthreads();
    if (tid < LL) w[tid] *= inv;
    __syncthreads();

    const float* eb = enc + (size_t)b * LL * HH;
    for (int n = tid; n < HH; n += 128) {
        float a = 0.f;
        for (int l = 0; l < LL; ++l) a += w[l] * eb[(size_t)l * HH + n];
        out[(size_t)b * HH + n] = a;
    }
}

// ---------------------------------------------------------------------------
// q[b,a] = m[b,:] . out_W[a,:] + out_b[a]  (tiny: 64 x 18 x 1024)
// ---------------------------------------------------------------------------
__global__ void out_proj_kernel(const float* __restrict__ m,
                                const float* __restrict__ outW,
                                const float* __restrict__ outB,
                                float* __restrict__ q)
{
    const int b = blockIdx.x, a = threadIdx.x;
    if (a < AA) {
        const float* mb = m + (size_t)b * HH;
        const float* wr = outW + (size_t)a * HH;
        float s = outB[a];
        for (int n = 0; n < HH; ++n) s += mb[n] * wr[n];
        q[b * AA + a] = s;
    }
}

// ---------------------------------------------------------------------------
extern "C" void kernel_launch(void* const* d_in, const int* in_sizes, int n_in,
                              void* d_out, int out_size, void* d_ws, size_t ws_size,
                              hipStream_t stream)
{
    (void)in_sizes; (void)n_in; (void)out_size; (void)ws_size;

    const float* state    = (const float*)d_in[0];   // (T,B,DIN)
    const float* demo     = (const float*)d_in[1];   // (T,B,L,DIN); demo0 = slice t=0
    const int*   dlen     = (const int*)  d_in[2];   // (T,B); lengths = row 0
    const float* h0       = (const float*)d_in[3];
    const float* c0       = (const float*)d_in[4];
    const float* enc_Wih  = (const float*)d_in[5];   // (4H, DIN)
    const float* enc_Whh  = (const float*)d_in[6];   // (4H, H)
    const float* enc_bih  = (const float*)d_in[7];
    const float* enc_bhh  = (const float*)d_in[8];
    const float* attn_W   = (const float*)d_in[9];   // (1, 2H)
    const float* attn_b   = (const float*)d_in[10];
    const float* comb_W   = (const float*)d_in[11];  // (H, H+DIN)
    const float* comb_b   = (const float*)d_in[12];
    const float* lstm_Wih = (const float*)d_in[13];  // (4H, H)
    const float* lstm_Whh = (const float*)d_in[14];  // (4H, H)
    const float* lstm_bih = (const float*)d_in[15];
    const float* lstm_bhh = (const float*)d_in[16];
    const float* mid_W    = (const float*)d_in[17];  // (H, H)
    const float* mid_b    = (const float*)d_in[18];
    const float* out_W    = (const float*)d_in[19];  // (A, H)
    const float* out_b    = (const float*)d_in[20];

    // ---- workspace suballocation (~52 MB) --------------------------------
    char* ws = (char*)d_ws;
    size_t off = 0;
    auto alloc = [&](size_t bytes) -> char* {
        char* p = ws + off; off += (bytes + 255) & ~(size_t)255; return p;
    };
    __hip_bfloat16* encWihP  = (__hip_bfloat16*)alloc((size_t)4 * HH * DIN * 2);
    __hip_bfloat16* encWhhP  = (__hip_bfloat16*)alloc((size_t)4 * HH * HH  * 2);
    __hip_bfloat16* lstmWihP = (__hip_bfloat16*)alloc((size_t)4 * HH * HH  * 2);
    __hip_bfloat16* lstmWhhP = (__hip_bfloat16*)alloc((size_t)4 * HH * HH  * 2);
    __hip_bfloat16* combP    = (__hip_bfloat16*)alloc((size_t)HH * (HH + DIN) * 2);
    __hip_bfloat16* midP     = (__hip_bfloat16*)alloc((size_t)HH * HH * 2);
    float* encBuf  = (float*)alloc((size_t)BB * LL * HH * 4);
    float* scoreD  = (float*)alloc((size_t)BB * LL * 4);
    float* hbuf0   = (float*)alloc((size_t)BB * HH * 4);
    float* hbuf1   = (float*)alloc((size_t)BB * HH * 4);
    float* cbuf    = (float*)alloc((size_t)BB * HH * 4);
    float* attnBuf = (float*)alloc((size_t)BB * HH * 4);
    float* xBuf    = (float*)alloc((size_t)BB * HH * 4);
    float* mBuf    = (float*)alloc((size_t)BB * HH * 4);

    // ---- weight packing (fp32 -> bf16 WMMA B-fragment layout) ------------
    pack_b_kernel<<<2048, 256, 0, stream>>>(enc_Wih,  encWihP,  DIN,      4 * HH);
    pack_b_kernel<<<2048, 256, 0, stream>>>(enc_Whh,  encWhhP,  HH,       4 * HH);
    pack_b_kernel<<<2048, 256, 0, stream>>>(lstm_Wih, lstmWihP, HH,       4 * HH);
    pack_b_kernel<<<2048, 256, 0, stream>>>(lstm_Whh, lstmWhhP, HH,       4 * HH);
    pack_b_kernel<<<2048, 256, 0, stream>>>(comb_W,   combP,    HH + DIN, HH);
    pack_b_kernel<<<2048, 256, 0, stream>>>(mid_W,    midP,     HH,       HH);

    // ---- encoder LSTM over demo0 (L=80 sequential steps) -----------------
    hipMemsetAsync(hbuf0, 0, (size_t)BB * HH * 4, stream);
    hipMemsetAsync(cbuf,  0, (size_t)BB * HH * 4, stream);
    float* hcur = hbuf0;
    float* hnxt = hbuf1;
    for (int t = 0; t < LL; ++t) {
        lstm_step_kernel<<<16, GTHREADS, 0, stream>>>(
            demo + (size_t)t * DIN, LL * DIN, DIN,
            hcur, encWihP, encWhhP, enc_bih, enc_bhh,
            hnxt, cbuf, encBuf + (size_t)t * HH, LL * HH);
        float* tmp = hcur; hcur = hnxt; hnxt = tmp;
    }

    score_d_kernel<<<BB * LL, 128, 0, stream>>>(encBuf, attn_W, attn_b, scoreD);

    // ---- decoder scan (T=40 sequential steps) ----------------------------
    hipMemcpyAsync(hbuf0, h0, (size_t)BB * HH * 4, hipMemcpyDeviceToDevice, stream);
    hipMemcpyAsync(cbuf,  c0, (size_t)BB * HH * 4, hipMemcpyDeviceToDevice, stream);
    hcur = hbuf0; hnxt = hbuf1;
    float* qbase = (float*)d_out;
    for (int t = 0; t < TT; ++t) {
        attn_kernel<<<BB, 128, 0, stream>>>(hcur, encBuf, scoreD, attn_W, dlen, attnBuf);
        gemm2_kernel<true><<<16, GTHREADS, 0, stream>>>(
            attnBuf, HH, HH,
            state + (size_t)t * BB * DIN, DIN, DIN,
            combP, comb_b, xBuf);
        lstm_step_kernel<<<16, GTHREADS, 0, stream>>>(
            xBuf, HH, HH,
            hcur, lstmWihP, lstmWhhP, lstm_bih, lstm_bhh,
            hnxt, cbuf, nullptr, 0);
        gemm2_kernel<false><<<16, GTHREADS, 0, stream>>>(
            hnxt, HH, HH, nullptr, 0, 0, midP, mid_b, mBuf);
        out_proj_kernel<<<BB, 32, 0, stream>>>(mBuf, out_W, out_b,
                                               qbase + (size_t)t * BB * AA);
        float* tmp = hcur; hcur = hnxt; hnxt = tmp;
    }

    // ---- final (h, c) outputs --------------------------------------------
    hipMemcpyAsync(qbase + (size_t)TT * BB * AA,            hcur,
                   (size_t)BB * HH * 4, hipMemcpyDeviceToDevice, stream);
    hipMemcpyAsync(qbase + (size_t)TT * BB * AA + BB * HH,  cbuf,
                   (size_t)BB * HH * 4, hipMemcpyDeviceToDevice, stream);
}
